// BlockSparseAttention_9131100471588
// MI455X (gfx1250) — matrix operand order repaired and checked by reading, not compiled, not run on hardware
//
#include <hip/hip_runtime.h>

// ---------- types ----------
typedef __attribute__((ext_vector_type(16))) __bf16 v16bf;
typedef __attribute__((ext_vector_type(8)))  __bf16 v8bf;
typedef __attribute__((ext_vector_type(8)))  float  v8f;

union AF { v16bf v; v8bf h[2]; };

static __device__ __forceinline__ unsigned short f2b(float f) {
    unsigned int u = __float_as_uint(f);
    u += 0x7fffu + ((u >> 16) & 1u);            // round-to-nearest-even
    return (unsigned short)(u >> 16);
}

static __device__ __forceinline__ v8f wmma_bf16(v16bf a, v16bf b, v8f c) {
    return __builtin_amdgcn_wmma_f32_16x16x32_bf16(false, a, false, b, (short)0, c,
                                                   false, false);
}

// ---------- CDNA5 async global->LDS copy (ASYNCcnt-tracked) ----------
// INST_OFFSET is added to BOTH the LDS and global addresses (ISA 08 §4.4),
// so lockstep-advancing copies share one base address with offset immediates.
static __device__ __forceinline__ unsigned lds_off(const void* p) {
    return (unsigned)(size_t)p;   // flat shared ptr: addr[31:0] = LDS byte offset
}
template <int IMM>
static __device__ __forceinline__ void async_ld_b128(unsigned lds_byte, const void* g) {
    asm volatile("global_load_async_to_lds_b128 %0, %1, off offset:%2"
                 :: "v"(lds_byte), "v"(g), "i"(IMM) : "memory");
}
static __device__ __forceinline__ void wait_async0() {
    asm volatile("s_wait_asynccnt 0x0" ::: "memory");
}

// ---------- fp32 -> bf16 convert ----------
__global__ __launch_bounds__(256) void cvt_f32_bf16(const float* __restrict__ in,
                                                    unsigned short* __restrict__ out,
                                                    int n) {
    int i = (blockIdx.x * 256 + threadIdx.x) * 4;
    if (i + 3 < n) {
        float4 f = *(const float4*)(in + i);
        unsigned short o[4] = { f2b(f.x), f2b(f.y), f2b(f.z), f2b(f.w) };
        *(uint2*)(out + i) = *(const uint2*)o;
    }
}

// ---------- fp32 [K][N] -> bf16 transposed [N][K] ----------
__global__ __launch_bounds__(256) void cvt_transpose_f32_bf16(
    const float* __restrict__ in, unsigned short* __restrict__ out,
    int Kdim, int Ndim) {
    __shared__ float tile[32][33];
    const int k0 = blockIdx.y * 32, n0 = blockIdx.x * 32;
    const int tx = threadIdx.x & 31, ty = threadIdx.x >> 5;   // ty 0..7
#pragma unroll
    for (int i = 0; i < 4; ++i) {
        const int k = ty + i * 8;
        tile[k][tx] = in[(size_t)(k0 + k) * Ndim + n0 + tx];
    }
    __syncthreads();
#pragma unroll
    for (int i = 0; i < 4; ++i) {
        const int n = ty + i * 8;
        out[(size_t)(n0 + n) * Kdim + k0 + tx] = f2b(tile[tx][n]);
    }
}

// ---------- GEMM: C = A[M,K] @ W[K,N] + bias,  W given pre-transposed Wt[N][K] ----
// WG tile 128x128, 8 waves (4x2), wave tile 32x64. Async double-buffered K staging.
template <bool OUTF32, bool TRANSOUT>
__global__ __launch_bounds__(256) void gemm_bias(
    const unsigned short* __restrict__ A,    // [M,K] bf16
    const unsigned short* __restrict__ Wt,   // [N,K] bf16 (pre-transposed)
    const float* __restrict__ bias,          // [N]
    void* __restrict__ Cout, int M, int N, int K)
{
    __shared__ __align__(16) unsigned short As[2][128][40];  // [m][k], pad 8
    __shared__ __align__(16) unsigned short Bs[2][128][40];  // [n][k], pad 8

    const int t    = threadIdx.x;
    const int lane = t & 31;
    const int w    = t >> 5;
    const int wm   = w >> 1;          // 0..3
    const int wn   = w & 1;           // 0..1
    const int ln   = lane & 15;
    const int hi   = lane >> 4;
    const int m_tile = blockIdx.y * 128;
    const int n_tile = blockIdx.x * 128;

    v8f acc[2][4];
#pragma unroll
    for (int i = 0; i < 2; ++i)
#pragma unroll
        for (int j = 0; j < 4; ++j) acc[i][j] = {};

    const int sr = t >> 1;            // stage row 0..127
    const int sk = (t & 1) * 16;      // k half (0|16)
    const unsigned short* Ag = A  + (size_t)(m_tile + sr) * K + sk;
    const unsigned short* Bg = Wt + (size_t)(n_tile + sr) * K + sk;
    const unsigned ldsA[2] = { lds_off(&As[0][sr][sk]), lds_off(&As[1][sr][sk]) };
    const unsigned ldsB[2] = { lds_off(&Bs[0][sr][sk]), lds_off(&Bs[1][sr][sk]) };

    // prologue: stage tile 0 into buffer 0
    async_ld_b128<0 >(ldsA[0], Ag);
    async_ld_b128<16>(ldsA[0], Ag);
    async_ld_b128<0 >(ldsB[0], Bg);
    async_ld_b128<16>(ldsB[0], Bg);
    Ag += 32; Bg += 32;

    const int niter = K >> 5;
    for (int it = 0; it < niter; ++it) {
        const int buf = it & 1;
        wait_async0();
        __syncthreads();
        if (it + 1 < niter) {
            async_ld_b128<0 >(ldsA[buf ^ 1], Ag);
            async_ld_b128<16>(ldsA[buf ^ 1], Ag);
            async_ld_b128<0 >(ldsB[buf ^ 1], Bg);
            async_ld_b128<16>(ldsB[buf ^ 1], Bg);
            Ag += 32; Bg += 32;
        }

        AF af[2];
#pragma unroll
        for (int i = 0; i < 2; ++i) {
            const unsigned short* p = &As[buf][wm * 32 + i * 16 + ln][0];
            af[i].h[0] = *(const v8bf*)(p + hi * 16);       // K = hi*16 + 0..7
            af[i].h[1] = *(const v8bf*)(p + hi * 16 + 8);   // K = hi*16 + 8..15
        }
#pragma unroll
        for (int j = 0; j < 4; ++j) {
            AF bm;
            const unsigned short* p = &Bs[buf][wn * 64 + j * 16 + ln][0];
            bm.h[0] = *(const v8bf*)(p + hi * 16);          // K = hi*16 + 0..7
            bm.h[1] = *(const v8bf*)(p + hi * 16 + 8);      // K = hi*16 + 8..15
#pragma unroll
            for (int i = 0; i < 2; ++i)
                acc[i][j] = wmma_bf16(af[i].v, bm.v, acc[i][j]);
        }
    }

    // epilogue (C layout: M = r + 8*hi, N = ln)
#pragma unroll
    for (int j = 0; j < 4; ++j) {
        const int ncol = n_tile + wn * 64 + j * 16 + ln;
        const float bv = bias[ncol];
#pragma unroll
        for (int i = 0; i < 2; ++i) {
            const int mrow0 = m_tile + wm * 32 + i * 16 + hi * 8;
            if (TRANSOUT) {
                // 8 consecutive rows at one column -> one 16B store to Ct[N][M]
                unsigned short pk[8];
#pragma unroll
                for (int r = 0; r < 8; ++r) pk[r] = f2b(acc[i][j][r] + bv);
                *(uint4*)((unsigned short*)Cout + (size_t)ncol * M + mrow0) =
                    *(const uint4*)pk;
            } else {
#pragma unroll
                for (int r = 0; r < 8; ++r) {
                    float v = acc[i][j][r] + bv;
                    if (OUTF32)
                        ((float*)Cout)[(size_t)(mrow0 + r) * N + ncol] = v;
                    else
                        ((unsigned short*)Cout)[(size_t)(mrow0 + r) * N + ncol] = f2b(v);
                }
            }
        }
    }
}

// ---------- block-local attention: one WG per (b, h, block) ----------
// Q,K: bf16 [B*S, 1024]; Vt: bf16 [1024, B*S] (transposed); Ctx: bf16 [B*S, 1024].
__global__ __launch_bounds__(256) void attn_block(
    const unsigned short* __restrict__ Qg,
    const unsigned short* __restrict__ Kg,
    const unsigned short* __restrict__ Vtg,
    unsigned short* __restrict__ Ctx,
    int Mtok)
{
    // union pool: [Qs 128x72 | Ks 128x72] overlaid later by [Ps 128x136]; Vt 64x136
    __shared__ __align__(16) unsigned short smem[27136];
    unsigned short* Qs = smem;             // pitch 72
    unsigned short* Ks = smem + 9216;      // pitch 72
    unsigned short* Ps = smem;             // pitch 136 (reuses Q/K region)
    unsigned short* Vt = smem + 18432;     // pitch 136, [d][token]

    const int t    = threadIdx.x;
    const int lane = t & 31;
    const int w    = t >> 5;               // wave 0..7 -> 16-row strip
    const int ln   = lane & 15;
    const int hi   = lane >> 4;
    const int blk  = blockIdx.x;
    const int h    = blockIdx.y;
    const int b    = blockIdx.z;
    const size_t tok0 = (size_t)b * 4096 + (size_t)blk * 128;
    const int colbase = h * 64;

    // ---- async stage Q, K rows and Vt (already transposed in global) ----
    {
        const int r    = t >> 1;
        const int half = (t & 1) * 32;
        const unsigned short* qsrc = Qg + (tok0 + r) * 1024 + colbase + half;
        const unsigned short* ksrc = Kg + (tok0 + r) * 1024 + colbase + half;
        const unsigned lq = lds_off(&Qs[r * 72 + half]);
        const unsigned lk = lds_off(&Ks[r * 72 + half]);
        async_ld_b128<0 >(lq, qsrc);  async_ld_b128<16>(lq, qsrc);
        async_ld_b128<32>(lq, qsrc);  async_ld_b128<48>(lq, qsrc);
        async_ld_b128<0 >(lk, ksrc);  async_ld_b128<16>(lk, ksrc);
        async_ld_b128<32>(lk, ksrc);  async_ld_b128<48>(lk, ksrc);

        const int d   = t >> 2;            // 0..63
        const int seg = (t & 3) * 32;      // token segment
        const unsigned short* vsrc = Vtg + (size_t)(colbase + d) * Mtok + tok0 + seg;
        const unsigned lv = lds_off(&Vt[d * 136 + seg]);
        async_ld_b128<0 >(lv, vsrc);  async_ld_b128<16>(lv, vsrc);
        async_ld_b128<32>(lv, vsrc);  async_ld_b128<48>(lv, vsrc);
    }
    wait_async0();
    __syncthreads();

    // ---- scores S = Q @ K^T : wave strip rows m0..m0+15, all 128 cols ----
    const int m0 = w * 16;
    v8f sacc[8];
#pragma unroll
    for (int j = 0; j < 8; ++j) sacc[j] = {};

#pragma unroll
    for (int k0 = 0; k0 < 64; k0 += 32) {
        AF a;
        const unsigned short* ap = Qs + (m0 + ln) * 72 + k0;
        a.h[0] = *(const v8bf*)(ap + hi * 16);
        a.h[1] = *(const v8bf*)(ap + hi * 16 + 8);
#pragma unroll
        for (int j = 0; j < 8; ++j) {
            AF bm;   // B = K^T -> element jj = K[n][k0 + hi*16 + jj]
            const unsigned short* bp = Ks + (j * 16 + ln) * 72 + k0 + hi * 16;
            bm.h[0] = *(const v8bf*)(bp);
            bm.h[1] = *(const v8bf*)(bp + 8);
            sacc[j] = wmma_bf16(a.v, bm.v, sacc[j]);
        }
    }
    __syncthreads();   // everyone done reading Qs/Ks; Ps may overwrite them

    // ---- softmax in registers on C layout (row m = m0 + r + 8*hi, col = j*16+ln)
    const float cc = 0.125f * 1.44269504088896340736f;   // 1/sqrt(64) * log2(e)
#pragma unroll
    for (int r = 0; r < 8; ++r) {
        float vals[8];
        float lm = -__builtin_inff();
#pragma unroll
        for (int j = 0; j < 8; ++j) { vals[j] = sacc[j][r]; lm = fmaxf(lm, vals[j]); }
#pragma unroll
        for (int msk = 1; msk < 16; msk <<= 1) lm = fmaxf(lm, __shfl_xor(lm, msk, 32));
        float ls = 0.f;
#pragma unroll
        for (int j = 0; j < 8; ++j) {
            vals[j] = __builtin_exp2f((vals[j] - lm) * cc);
            ls += vals[j];
        }
#pragma unroll
        for (int msk = 1; msk < 16; msk <<= 1) ls += __shfl_xor(ls, msk, 32);
        const float inv = 1.0f / ls;
        const int mrow = m0 + r + hi * 8;
#pragma unroll
        for (int j = 0; j < 8; ++j)
            Ps[mrow * 136 + j * 16 + ln] = f2b(vals[j] * inv);
    }
    __syncthreads();

    // ---- O = P @ V : wave strip rows m0..m0+15, 64 cols ----
    v8f oacc[4];
#pragma unroll
    for (int j = 0; j < 4; ++j) oacc[j] = {};

#pragma unroll
    for (int k0 = 0; k0 < 128; k0 += 32) {
        AF a;
        const unsigned short* ap = Ps + (m0 + ln) * 136 + k0;
        a.h[0] = *(const v8bf*)(ap + hi * 16);
        a.h[1] = *(const v8bf*)(ap + hi * 16 + 8);
#pragma unroll
        for (int j = 0; j < 4; ++j) {
            AF bm;   // B = V -> element jj = Vt[n][k0 + hi*16 + jj]
            const unsigned short* bp = Vt + (j * 16 + ln) * 136 + k0 + hi * 16;
            bm.h[0] = *(const v8bf*)(bp);
            bm.h[1] = *(const v8bf*)(bp + 8);
            oacc[j] = wmma_bf16(a.v, bm.v, oacc[j]);
        }
    }

    // store context (bf16) at [token, h*64 + n]
#pragma unroll
    for (int j = 0; j < 4; ++j) {
        const int col = colbase + j * 16 + ln;
#pragma unroll
        for (int r = 0; r < 8; ++r) {
            const size_t row = tok0 + m0 + r + hi * 8;
            Ctx[row * 1024 + col] = f2b(oacc[j][r]);
        }
    }
}

// ---------- host launch ----------
extern "C" void kernel_launch(void* const* d_in, const int* in_sizes, int n_in,
                              void* d_out, int out_size, void* d_ws, size_t ws_size,
                              hipStream_t stream) {
    (void)in_sizes; (void)n_in; (void)out_size; (void)ws_size;
    const float* x  = (const float*)d_in[0];
    const float* Wq = (const float*)d_in[1];
    const float* bq = (const float*)d_in[2];
    const float* Wk = (const float*)d_in[3];
    const float* bk = (const float*)d_in[4];
    const float* Wv = (const float*)d_in[5];
    const float* bv = (const float*)d_in[6];
    const float* Wo = (const float*)d_in[7];
    const float* bo = (const float*)d_in[8];
    float* out = (float*)d_out;

    const int Mrows = 4 * 4096;      // B*S
    const int D = 1024;

    unsigned short* ws = (unsigned short*)d_ws;
    size_t off = 0;
    unsigned short* Xb  = ws + off; off += (size_t)Mrows * D;
    unsigned short* Wqt = ws + off; off += (size_t)D * D;
    unsigned short* Wkt = ws + off; off += (size_t)D * D;
    unsigned short* Wvt = ws + off; off += (size_t)D * D;
    unsigned short* Wot = ws + off; off += (size_t)D * D;
    unsigned short* Qb  = ws + off; off += (size_t)Mrows * D;
    unsigned short* Kb  = ws + off; off += (size_t)Mrows * D;
    unsigned short* Vbt = ws + off; off += (size_t)Mrows * D;   // [D][Mrows]
    unsigned short* Cb  = ws + off; off += (size_t)Mrows * D;

    // x: fp32 -> bf16 ; weights: fp32 [K][N] -> bf16 transposed [N][K]
    cvt_f32_bf16<<<(Mrows * D) / (4 * 256), 256, 0, stream>>>(x, Xb, Mrows * D);
    dim3 gt(D / 32, D / 32);
    cvt_transpose_f32_bf16<<<gt, 256, 0, stream>>>(Wq, Wqt, D, D);
    cvt_transpose_f32_bf16<<<gt, 256, 0, stream>>>(Wk, Wkt, D, D);
    cvt_transpose_f32_bf16<<<gt, 256, 0, stream>>>(Wv, Wvt, D, D);
    cvt_transpose_f32_bf16<<<gt, 256, 0, stream>>>(Wo, Wot, D, D);

    // projections (V stored transposed for the attention B-operand)
    dim3 gg(D / 128, Mrows / 128);
    gemm_bias<false, false><<<gg, 256, 0, stream>>>(Xb, Wqt, bq, (void*)Qb,  Mrows, D, D);
    gemm_bias<false, false><<<gg, 256, 0, stream>>>(Xb, Wkt, bk, (void*)Kb,  Mrows, D, D);
    gemm_bias<false, true ><<<gg, 256, 0, stream>>>(Xb, Wvt, bv, (void*)Vbt, Mrows, D, D);

    // block-local attention
    dim3 ga(32, 16, 4);   // (block, head, batch)
    attn_block<<<ga, 256, 0, stream>>>(Qb, Kb, Vbt, Cb, Mrows);

    // output projection -> f32
    gemm_bias<true, false><<<gg, 256, 0, stream>>>(Cb, Wot, bo, (void*)out, Mrows, D, D);
}
